// FLASH_6073083756839
// MI455X (gfx1250) — matrix operand order repaired and checked by reading, not compiled, not run on hardware
//
#include <hip/hip_runtime.h>
#include <hip/hip_bf16.h>
#include <cstdint>

typedef __attribute__((ext_vector_type(16))) __bf16 v16bf;
typedef __attribute__((ext_vector_type(8)))  float  v8f;

#define BM 128
#define BN 128
#define BK 32
#define LDAS 40   // ushorts per LDS A row (64B data + 16B pad, keeps 16B alignment)
#define LDBS 40
#define NTHREADS 256

static constexpr int H_ = 1024;
static constexpr int E_ = 2048;
static constexpr int SD = 128;          // S_DIM
static constexpr int Bb = 4;
static constexpr int Ss = 2048;
static constexpr int F_ = 2 * E_ + SD;  // 4224

struct alignas(16) U4 { unsigned int x, y, z, w; };
struct alignas(8)  U2 { unsigned int x, y; };

__device__ __forceinline__ unsigned short f2bf(float f) {
  unsigned int u = __float_as_uint(f);
  unsigned int r = u + 0x7FFFu + ((u >> 16) & 1u);   // round-to-nearest-even
  return (unsigned short)(r >> 16);
}
__device__ __forceinline__ float bf2f(unsigned short h) {
  return __uint_as_float(((unsigned int)h) << 16);
}
__device__ __forceinline__ float fast_silu(float x) {
  // bf16-accuracy SiLU: x * rcp(1+exp(-x)) with single v_rcp_f32
  return x * __builtin_amdgcn_rcpf(1.0f + __expf(-x));
}
// low 32 bits of a flat LDS pointer == byte offset within the wave's LDS allocation
__device__ __forceinline__ unsigned lds_off(const void* p) {
  return (unsigned)(uintptr_t)p;
}
// async copy 16B global -> LDS (ASYNCcnt-tracked, bypasses VGPRs)
__device__ __forceinline__ void async_b128(unsigned loff, unsigned goff, const void* base) {
  asm volatile("global_load_async_to_lds_b128 %0, %1, %2"
               :: "v"(loff), "v"(goff), "s"(base) : "memory");
}

union FragAB { v16bf v; unsigned int u[8]; };

// One K=32 step for a wave's 64x32 sub-tile: 4 A-frags, 2 B-frags, 8 WMMAs.
// As: M-major [BM][LDAS]; Bs: N-major (i.e. B transposed) [BN][LDBS].
__device__ __forceinline__ void mma_tile(const unsigned short* As,
                                         const unsigned short* Bs,
                                         int waveM, int waveN, int lane,
                                         v8f acc[4][2]) {
  const int half = lane >> 4;   // 0: lanes 0-15, 1: lanes 16-31
  const int l15  = lane & 15;
  FragAB fa[4];
#pragma unroll
  for (int mi = 0; mi < 4; ++mi) {
    const unsigned short* p = As + (waveM * 64 + mi * 16 + l15) * LDAS;
    // lanes 0-15: K 0..7 and 16..23 ; lanes 16-31: K 8..15 and 24..31
    *(U4*)&fa[mi].u[0] = *(const U4*)(p + half * 8);
    *(U4*)&fa[mi].u[4] = *(const U4*)(p + 16 + half * 8);
  }
  FragAB fb[2];
#pragma unroll
  for (int ni = 0; ni < 2; ++ni) {
    const unsigned short* p = Bs + (waveN * 32 + ni * 16 + l15) * LDBS;
    // lanes 0-15: K 0..7 and 16..23 ; lanes 16-31: K 8..15 and 24..31
    *(U4*)&fb[ni].u[0] = *(const U4*)(p + half * 8);
    *(U4*)&fb[ni].u[4] = *(const U4*)(p + 16 + half * 8);
  }
#pragma unroll
  for (int mi = 0; mi < 4; ++mi)
#pragma unroll
    for (int ni = 0; ni < 2; ++ni)
      acc[mi][ni] = __builtin_amdgcn_wmma_f32_16x16x32_bf16(
          false, fa[mi].v, false, fb[ni].v, (short)0, acc[mi][ni], false, false);
}

// ---------------- Kernel 1: uvb = silu(X @ W_uv + b_uv); split into u,v,(q,k) ----
__global__ __launch_bounds__(NTHREADS)
void k_uvb(const float* __restrict__ X, const float* __restrict__ Wuv,
           const float* __restrict__ buv, const float* __restrict__ gamma,
           const float* __restrict__ beta,
           unsigned short* __restrict__ u_bf, unsigned short* __restrict__ v_bf,
           unsigned short* __restrict__ q_bf, unsigned short* __restrict__ k_bf) {
  __shared__ __align__(16) unsigned short As[2][BM * LDAS];
  __shared__ __align__(16) unsigned short Bs[2][BN * LDBS];
  const int tid = threadIdx.x, lane = tid & 31, wave = tid >> 5;
  const int waveM = wave >> 2, waveN = wave & 3;
  const int mBase = blockIdx.y * BM, nBase = blockIdx.x * BN;
  v8f acc[4][2] = {};

  float4 ra[4], rb[4];
  auto loadRegs = [&](int k0) {
#pragma unroll
    for (int i = 0; i < 4; ++i) {
      int idx = tid + i * NTHREADS;              // 0..1023
      int m = idx >> 3, k4 = (idx & 7) << 2;
      ra[i] = *(const float4*)(X + (size_t)(mBase + m) * H_ + k0 + k4);
    }
#pragma unroll
    for (int i = 0; i < 4; ++i) {
      int idx = tid + i * NTHREADS;
      int k = idx >> 5, n4 = (idx & 31) << 2;
      rb[i] = *(const float4*)(Wuv + (size_t)(k0 + k) * F_ + nBase + n4);
    }
  };
  auto storeLds = [&](int buf) {
#pragma unroll
    for (int i = 0; i < 4; ++i) {
      int idx = tid + i * NTHREADS;
      int m = idx >> 3, k4 = (idx & 7) << 2;
      U2 pk;
      pk.x = (unsigned)f2bf(ra[i].x) | ((unsigned)f2bf(ra[i].y) << 16);
      pk.y = (unsigned)f2bf(ra[i].z) | ((unsigned)f2bf(ra[i].w) << 16);
      *(U2*)(&As[buf][m * LDAS + k4]) = pk;
    }
#pragma unroll
    for (int i = 0; i < 4; ++i) {
      int idx = tid + i * NTHREADS;
      int k = idx >> 5, n4 = (idx & 31) << 2;
      Bs[buf][(n4 + 0) * LDBS + k] = f2bf(rb[i].x);
      Bs[buf][(n4 + 1) * LDBS + k] = f2bf(rb[i].y);
      Bs[buf][(n4 + 2) * LDBS + k] = f2bf(rb[i].z);
      Bs[buf][(n4 + 3) * LDBS + k] = f2bf(rb[i].w);
    }
  };

  constexpr int NS = H_ / BK;                    // 32
  loadRegs(0);
  storeLds(0);
  for (int step = 0; step < NS; ++step) {
    int cur = step & 1;
    if (step + 1 < NS) loadRegs((step + 1) * BK);  // overlap global latency w/ WMMA
    __syncthreads();
    mma_tile(As[cur], Bs[cur], waveM, waveN, lane, acc);
    __syncthreads();
    if (step + 1 < NS) storeLds(cur ^ 1);
  }

  const int half = lane >> 4, l15 = lane & 15;
#pragma unroll
  for (int ni = 0; ni < 2; ++ni) {
    int col = nBase + waveN * 32 + ni * 16 + l15;
    float bias = buv[col];
#pragma unroll
    for (int mi = 0; mi < 4; ++mi) {
#pragma unroll
      for (int r = 0; r < 8; ++r) {
        int row = mBase + waveM * 64 + mi * 16 + half * 8 + r;
        float s = fast_silu(acc[mi][ni][r] + bias);
        if (col < E_) {
          u_bf[(size_t)row * E_ + col] = f2bf(s);
        } else if (col < 2 * E_) {
          v_bf[(size_t)row * E_ + (col - E_)] = f2bf(s);
        } else {
          int j = col - 2 * E_;
          q_bf[(size_t)row * SD + j] = f2bf(s * gamma[j]      + beta[j]);
          k_bf[(size_t)row * SD + j] = f2bf(s * gamma[SD + j] + beta[SD + j]);
        }
      }
    }
  }
}

// ---------------- Kernel 2a: sq = relu(q k^T / S)^2 per batch --------------------
// Both operands are plain bf16 tile copies -> fully async global->LDS staged.
__global__ __launch_bounds__(NTHREADS)
void k_scores(const unsigned short* __restrict__ q_bf,
              const unsigned short* __restrict__ k_bf,
              unsigned short* __restrict__ sq_bf) {
  __shared__ __align__(16) unsigned short As[2][BM * LDAS];
  __shared__ __align__(16) unsigned short Bs[2][BN * LDBS];
  const int tid = threadIdx.x, lane = tid & 31, wave = tid >> 5;
  const int waveM = wave >> 2, waveN = wave & 3;
  const int b = blockIdx.z;
  const int mBase = blockIdx.y * BM, nBase = blockIdx.x * BN;
  const unsigned short* qb = q_bf + (size_t)b * Ss * SD;
  const unsigned short* kb = k_bf + (size_t)b * Ss * SD;
  v8f acc[4][2] = {};

  auto issueTile = [&](int buf, int k0) {        // 4 async instrs / wave / tile
#pragma unroll
    for (int i = 0; i < 2; ++i) {
      int idx = tid + i * NTHREADS;              // 0..511
      int m = idx >> 2, kq = idx & 3;
      async_b128(lds_off(&As[buf][m * LDAS + kq * 8]),
                 (unsigned)(((mBase + m) * SD + k0) * 2 + kq * 16), qb);
    }
#pragma unroll
    for (int i = 0; i < 2; ++i) {
      int idx = tid + i * NTHREADS;
      int n = idx >> 2, kq = idx & 3;
      async_b128(lds_off(&Bs[buf][n * LDBS + kq * 8]),
                 (unsigned)(((nBase + n) * SD + k0) * 2 + kq * 16), kb);
    }
  };

  constexpr int NS = SD / BK;                    // 4
  issueTile(0, 0);
  for (int step = 0; step < NS; ++step) {
    int cur = step & 1;
    if (step + 1 < NS) {
      issueTile(cur ^ 1, (step + 1) * BK);
      asm volatile("s_wait_asynccnt 4" ::: "memory");   // tile `step` landed
    } else {
      asm volatile("s_wait_asynccnt 0" ::: "memory");
    }
    __syncthreads();
    mma_tile(As[cur], Bs[cur], waveM, waveN, lane, acc);
    __syncthreads();
  }

  const int half = lane >> 4, l15 = lane & 15;
  const float invS = 1.0f / (float)Ss;
  unsigned short* dst = sq_bf + (size_t)b * Ss * Ss;
#pragma unroll
  for (int ni = 0; ni < 2; ++ni) {
    int col = nBase + waveN * 32 + ni * 16 + l15;
#pragma unroll
    for (int mi = 0; mi < 4; ++mi)
#pragma unroll
      for (int r = 0; r < 8; ++r) {
        int row = mBase + waveM * 64 + mi * 16 + half * 8 + r;
        float s = acc[mi][ni][r] * invS;
        float rl = fmaxf(s, 0.0f);
        dst[(size_t)row * Ss + col] = f2bf(rl * rl);
      }
  }
}

// ---------------- Kernel 2b: h = u * (sq @ v) per batch --------------------------
// A (sq tile): async copy. B (v tile): needs transpose -> register-pipelined.
__global__ __launch_bounds__(NTHREADS)
void k_attnv(const unsigned short* __restrict__ sq_bf,
             const unsigned short* __restrict__ v_bf,
             const unsigned short* __restrict__ u_bf,
             unsigned short* __restrict__ h_bf) {
  __shared__ __align__(16) unsigned short As[2][BM * LDAS];
  __shared__ __align__(16) unsigned short Bs[2][BN * LDBS];
  const int tid = threadIdx.x, lane = tid & 31, wave = tid >> 5;
  const int waveM = wave >> 2, waveN = wave & 3;
  const int b = blockIdx.z;
  const int mBase = blockIdx.y * BM, nBase = blockIdx.x * BN;
  const unsigned short* sqb = sq_bf + (size_t)b * Ss * Ss;
  v8f acc[4][2] = {};

  auto issueA = [&](int buf, int k0) {           // 2 async instrs / wave / tile
#pragma unroll
    for (int i = 0; i < 2; ++i) {
      int idx = tid + i * NTHREADS;
      int m = idx >> 2, kq = idx & 3;
      async_b128(lds_off(&As[buf][m * LDAS + kq * 8]),
                 (unsigned)(((mBase + m) * Ss + k0) * 2 + kq * 16), sqb);
    }
  };
  unsigned wv[8];
  auto loadB = [&](int k0) {
#pragma unroll
    for (int i = 0; i < 8; ++i) {
      int di = tid + i * NTHREADS;               // 0..2047 dwords of 32x128 tile
      int k = di >> 6, nd = (di & 63) << 1;      // coalesced along n
      wv[i] = *(const unsigned int*)(v_bf + (size_t)(b * Ss + k0 + k) * E_ + nBase + nd);
    }
  };
  auto storeB = [&](int buf) {                   // transpose into Bs[n][k]
#pragma unroll
    for (int i = 0; i < 8; ++i) {
      int di = tid + i * NTHREADS;
      int k = di >> 6, nd = (di & 63) << 1;
      Bs[buf][(nd + 0) * LDBS + k] = (unsigned short)(wv[i] & 0xFFFFu);
      Bs[buf][(nd + 1) * LDBS + k] = (unsigned short)(wv[i] >> 16);
    }
  };

  constexpr int NS = Ss / BK;                    // 64
  issueA(0, 0);
  loadB(0);
  storeB(0);
  for (int step = 0; step < NS; ++step) {
    int cur = step & 1;
    if (step + 1 < NS) {
      issueA(cur ^ 1, (step + 1) * BK);
      loadB((step + 1) * BK);
      asm volatile("s_wait_asynccnt 2" ::: "memory");
    } else {
      asm volatile("s_wait_asynccnt 0" ::: "memory");
    }
    __syncthreads();
    mma_tile(As[cur], Bs[cur], waveM, waveN, lane, acc);
    __syncthreads();
    if (step + 1 < NS) storeB(cur ^ 1);
  }

  const int half = lane >> 4, l15 = lane & 15;
#pragma unroll
  for (int ni = 0; ni < 2; ++ni) {
    int col = nBase + waveN * 32 + ni * 16 + l15;
#pragma unroll
    for (int mi = 0; mi < 4; ++mi)
#pragma unroll
      for (int r = 0; r < 8; ++r) {
        int row = mBase + waveM * 64 + mi * 16 + half * 8 + r;
        size_t g = (size_t)(b * Ss + row) * E_ + col;
        h_bf[g] = f2bf(acc[mi][ni][r] * bf2f(u_bf[g]));
      }
  }
}

// ---------------- Kernel 3: out = h @ W_out + b_out ------------------------------
// A (h tile): async copy. B (W_out): f32->bf16 + transpose -> register-pipelined.
__global__ __launch_bounds__(NTHREADS)
void k_out(const unsigned short* __restrict__ h_bf,
           const float* __restrict__ Wout, const float* __restrict__ bout,
           float* __restrict__ out) {
  __shared__ __align__(16) unsigned short As[2][BM * LDAS];
  __shared__ __align__(16) unsigned short Bs[2][BN * LDBS];
  const int tid = threadIdx.x, lane = tid & 31, wave = tid >> 5;
  const int waveM = wave >> 2, waveN = wave & 3;
  const int mBase = blockIdx.y * BM, nBase = blockIdx.x * BN;
  v8f acc[4][2] = {};

  auto issueA = [&](int buf, int k0) {
#pragma unroll
    for (int i = 0; i < 2; ++i) {
      int idx = tid + i * NTHREADS;
      int m = idx >> 2, kq = idx & 3;
      async_b128(lds_off(&As[buf][m * LDAS + kq * 8]),
                 (unsigned)(((mBase + m) * E_ + k0) * 2 + kq * 16), h_bf);
    }
  };
  float4 rbv[4];
  auto loadB = [&](int k0) {
#pragma unroll
    for (int i = 0; i < 4; ++i) {
      int idx = tid + i * NTHREADS;
      int k = idx >> 5, n4 = (idx & 31) << 2;
      rbv[i] = *(const float4*)(Wout + (size_t)(k0 + k) * H_ + nBase + n4);
    }
  };
  auto storeB = [&](int buf) {
#pragma unroll
    for (int i = 0; i < 4; ++i) {
      int idx = tid + i * NTHREADS;
      int k = idx >> 5, n4 = (idx & 31) << 2;
      Bs[buf][(n4 + 0) * LDBS + k] = f2bf(rbv[i].x);
      Bs[buf][(n4 + 1) * LDBS + k] = f2bf(rbv[i].y);
      Bs[buf][(n4 + 2) * LDBS + k] = f2bf(rbv[i].z);
      Bs[buf][(n4 + 3) * LDBS + k] = f2bf(rbv[i].w);
    }
  };

  constexpr int NS = E_ / BK;                    // 64
  issueA(0, 0);
  loadB(0);
  storeB(0);
  for (int step = 0; step < NS; ++step) {
    int cur = step & 1;
    if (step + 1 < NS) {
      issueA(cur ^ 1, (step + 1) * BK);
      loadB((step + 1) * BK);
      asm volatile("s_wait_asynccnt 2" ::: "memory");
    } else {
      asm volatile("s_wait_asynccnt 0" ::: "memory");
    }
    __syncthreads();
    mma_tile(As[cur], Bs[cur], waveM, waveN, lane, acc);
    __syncthreads();
    if (step + 1 < NS) storeB(cur ^ 1);
  }

  const int half = lane >> 4, l15 = lane & 15;
#pragma unroll
  for (int ni = 0; ni < 2; ++ni) {
    int col = nBase + waveN * 32 + ni * 16 + l15;
    float bias = bout[col];
#pragma unroll
    for (int mi = 0; mi < 4; ++mi)
#pragma unroll
      for (int r = 0; r < 8; ++r) {
        int row = mBase + waveM * 64 + mi * 16 + half * 8 + r;
        out[(size_t)row * H_ + col] = acc[mi][ni][r] + bias;
      }
  }
}

extern "C" void kernel_launch(void* const* d_in, const int* in_sizes, int n_in,
                              void* d_out, int out_size, void* d_ws, size_t ws_size,
                              hipStream_t stream) {
  const float* X     = (const float*)d_in[0];
  const float* Wuv   = (const float*)d_in[1];
  const float* buv   = (const float*)d_in[2];
  const float* gamma = (const float*)d_in[3];
  const float* beta  = (const float*)d_in[4];
  const float* Wout  = (const float*)d_in[5];
  const float* bout  = (const float*)d_in[6];
  float* out = (float*)d_out;

  // Workspace layout (bytes):
  //   u_bf  : 8192x2048 bf16  =  32 MiB @ 0
  //   v_bf  : 8192x2048 bf16  =  32 MiB @ 32Mi
  //   q_bf  : 8192x128  bf16  =   2 MiB @ 64Mi
  //   k_bf  : 8192x128  bf16  =   2 MiB @ 66Mi
  //   sq_bf : 4x2048x2048 bf16 = 32 MiB @ 68Mi
  //   h_bf  : 8192x2048 bf16  =  32 MiB @ 100Mi   (total 132 MiB)
  char* ws = (char*)d_ws;
  unsigned short* u_bf  = (unsigned short*)(ws);
  unsigned short* v_bf  = (unsigned short*)(ws + (size_t)33554432);
  unsigned short* q_bf  = (unsigned short*)(ws + (size_t)67108864);
  unsigned short* k_bf  = (unsigned short*)(ws + (size_t)69206016);
  unsigned short* sq_bf = (unsigned short*)(ws + (size_t)71303168);
  unsigned short* h_bf  = (unsigned short*)(ws + (size_t)104857600);

  dim3 blk(NTHREADS);
  k_uvb  <<<dim3(F_ / BN, (Bb * Ss) / BM),      blk, 0, stream>>>(X, Wuv, buv, gamma, beta,
                                                                  u_bf, v_bf, q_bf, k_bf);
  k_scores<<<dim3(Ss / BN, Ss / BM, Bb),        blk, 0, stream>>>(q_bf, k_bf, sq_bf);
  k_attnv<<<dim3(E_ / BN, Ss / BM, Bb),         blk, 0, stream>>>(sq_bf, v_bf, u_bf, h_bf);
  k_out  <<<dim3(H_ / BN, (Bb * Ss) / BM),      blk, 0, stream>>>(h_bf, Wout, bout, out);
}
